// HybridHCNet3D_13950053777570
// MI455X (gfx1250) — matrix operand order repaired and checked by reading, not compiled, not run on hardware
//
#include <hip/hip_runtime.h>
#include <math.h>

#define BB   4
#define NN   2048
#define HH   128
#define LL   4
#define KK   16
#define NRBFC 20
#define MVC  8
#define GSZ  8
#define ROWS (BB*NN)
#define CUTF 5.0f

typedef __attribute__((ext_vector_type(16))) _Float16 v16h;
typedef __attribute__((ext_vector_type(8)))  float    v8f;

// ---------------- CDNA5 async copy: global -> LDS, 16 bytes per lane ----------------
__device__ __forceinline__ void async_cp16(unsigned lds_off, const void* gaddr) {
  asm volatile("global_load_async_to_lds_b128 %0, %1, off"
               :: "v"(lds_off), "v"(gaddr) : "memory");
}
__device__ __forceinline__ void async_wait0() {
  asm volatile("s_wait_asynccnt 0x0" ::: "memory");
}

// ---------------- WMMA fragment loaders (CDNA5 ISA 7.12.2 layouts) ----------------
__device__ __forceinline__ v16h load_a16(const _Float16* A, int lda, int kb) {
  int lane  = threadIdx.x & 31;
  int m     = lane & 15;
  int kbase = (lane >> 4) * 8;
  const _Float16* p = A + m * lda + kb;
  v16h a;
#pragma unroll
  for (int e = 0; e < 8; ++e) a[e] = p[kbase + e];
#pragma unroll
  for (int e = 0; e < 8; ++e) a[8 + e] = p[16 + kbase + e];
  return a;
}
__device__ __forceinline__ v16h load_bT(const _Float16* Bt, int ldk, int n0, int kb) {
  int lane = threadIdx.x & 31;
  const _Float16* p = Bt + (size_t)(n0 + (lane & 15)) * ldk + kb + ((lane >> 4) * 8);
  v16h b;
#pragma unroll
  for (int e = 0; e < 8; ++e) b[e] = p[e];
#pragma unroll
  for (int e = 0; e < 8; ++e) b[8 + e] = p[16 + e];
  return b;
}

// fast silu: single v_exp_f32 + single v_rcp_f32 (error far below f16 GEMM noise)
__device__ __forceinline__ float siluf(float x) {
  return x * __builtin_amdgcn_rcpf(1.f + __expf(-x));
}
__device__ __forceinline__ float geluf(float x) { return 0.5f * x * (1.f + erff(x * 0.70710678118654752f)); }

// ---------------- KNN + RBF(f16, padded to 32) + envelope: one wave32 per node ----------------
__global__ __launch_bounds__(32) void knn_kernel(const float* __restrict__ x,
                                                 int* __restrict__ idx,
                                                 _Float16* __restrict__ rbf16,
                                                 float* __restrict__ env) {
  int row  = blockIdx.x;
  int b    = row >> 11;
  int n    = row & (NN - 1);
  int lane = threadIdx.x;
  const float* xb = x + (size_t)b * NN * 6;
  float px = xb[n * 6 + 0], py = xb[n * 6 + 1], pz = xb[n * 6 + 2];
  float d2loc[64];
#pragma unroll 8
  for (int t = 0; t < 64; ++t) {
    int j = t * 32 + lane;
    float dx = px - xb[j * 6 + 0];
    float dy = py - xb[j * 6 + 1];
    float dz = pz - xb[j * 6 + 2];
    float d2 = dx * dx + dy * dy + dz * dz;
    if (j == n) d2 += 1e9f;
    d2loc[t] = d2;
  }
  unsigned long long used = 0ull;
  for (int r = 0; r < KK; ++r) {
    float best = 3.4e38f; int bt = 0;
    for (int t = 0; t < 64; ++t)
      if (!((used >> t) & 1ull) && d2loc[t] < best) { best = d2loc[t]; bt = t; }
    float md = best; int mj = bt * 32 + lane; int ml = lane;
    for (int off = 16; off >= 1; off >>= 1) {
      float od = __shfl_xor(md, off);
      int   oj = __shfl_xor(mj, off);
      int   ol = __shfl_xor(ml, off);
      if (od < md || (od == md && ol < ml)) { md = od; mj = oj; ml = ol; }
    }
    if (lane == ml) used |= (1ull << bt);
    if (lane == 0) {
      idx[row * KK + r] = mj;
      float d = sqrtf(fmaxf(md, 1e-12f));
      env[row * KK + r] = (d < CUTF) ? 0.5f * (cosf(3.14159265358979f * d / CUTF) + 1.f) : 0.f;
      for (int c = 0; c < 32; ++c) {
        float v = 0.f;
        if (c < NRBFC) {
          float center = CUTF * (float)c / (float)(NRBFC - 1);
          float tt = (d - center) / (CUTF / NRBFC);
          v = __expf(-tt * tt);
        }
        rbf16[(size_t)(row * KK + r) * 32 + c] = (_Float16)v;
      }
    }
  }
}

// ---------------- input embedding: h (f32) + f16 mirror ----------------
__global__ void embed_kernel(const float* __restrict__ x, const float* __restrict__ in_w,
                             const float* __restrict__ in_b, float* __restrict__ h,
                             _Float16* __restrict__ h16) {
  int q = blockIdx.x * 256 + threadIdx.x;
  if (q >= ROWS * HH) return;
  int row = q >> 7, c = q & 127;
  float acc = in_b[c];
#pragma unroll
  for (int k = 0; k < 6; ++k) acc += x[row * 6 + k] * in_w[k * HH + c];
  h[q] = acc;
  h16[q] = (_Float16)acc;
}

// ---------------- weight convert + transpose: dst[N][Kpad] f16 ----------------
__global__ void prep_kernel(const float* __restrict__ src, _Float16* __restrict__ dst,
                            int K, int N, int Kpad) {
  int q = blockIdx.x * 256 + threadIdx.x;
  if (q >= N * Kpad) return;
  int n = q / Kpad, k = q % Kpad;
  dst[q] = (k < K) ? (_Float16)src[k * N + n] : (_Float16)0.f;
}

// ---------------- mean-field + mmv per batch ----------------
__global__ __launch_bounds__(128) void meanfield_kernel(const float* __restrict__ h,
                                                        const float* __restrict__ mp_w,
                                                        const float* __restrict__ mp_b,
                                                        float* __restrict__ mmv) {
  __shared__ float mfs[HH];
  int b = blockIdx.x, c = threadIdx.x;
  const float* hb = h + (size_t)b * NN * HH;
  float s = 0.f;
  for (int nn = 0; nn < NN; ++nn) s += hb[nn * HH + c];
  mfs[c] = s / (float)NN;
  __syncthreads();
  if (c < MVC) {
    float a = mp_b[c];
    for (int k = 0; k < HH; ++k) a += mfs[k] * mp_w[k * MVC + c];
    mmv[b * MVC + c] = a;
  }
}

// ---------------- full layer: 16 nodes per 8-wave block ----------------
struct LayerArgs {
  const float* h_in; float* h_out;
  const _Float16* h16_in; _Float16* h16_out;
  const int* idx; const _Float16* rbf16; const float* env; const float* mmv;
  const _Float16 *w1t, *w2t, *nmt, *ipt, *fut, *fc1t, *fc2t;
  const float *e1b, *e2b, *nmb, *lng, *lnb, *ppw, *ppb, *ipb, *gng, *gnb;
  const float *fub, *fc1b, *fc2b, *gmw, *gmb, *n1g, *n1b, *n2g, *n2b;
};

__device__ __forceinline__ void ln_row_f32(float* row, const float* g, const float* b) {
  float m = 0.f;
  for (int c = 0; c < HH; ++c) m += row[c];
  m *= (1.f / HH);
  float v = 0.f;
  for (int c = 0; c < HH; ++c) { float d = row[c] - m; v += d * d; }
  v *= (1.f / HH);
  float inv = rsqrtf(v + 1e-5f);
  for (int c = 0; c < HH; ++c) row[c] = (row[c] - m) * inv * g[c] + b[c];
}
__device__ __forceinline__ void ln_row_to_h(const float* row, _Float16* dst,
                                            const float* g, const float* b) {
  float m = 0.f;
  for (int c = 0; c < HH; ++c) m += row[c];
  m *= (1.f / HH);
  float v = 0.f;
  for (int c = 0; c < HH; ++c) { float d = row[c] - m; v += d * d; }
  v *= (1.f / HH);
  float inv = rsqrtf(v + 1e-5f);
  for (int c = 0; c < HH; ++c) dst[c] = (_Float16)((row[c] - m) * inv * g[c] + b[c]);
}
__device__ __forceinline__ void ln_row_out(const float* row, float* dst, _Float16* dsth,
                                           const float* g, const float* b) {
  float m = 0.f;
  for (int c = 0; c < HH; ++c) m += row[c];
  m *= (1.f / HH);
  float v = 0.f;
  for (int c = 0; c < HH; ++c) { float d = row[c] - m; v += d * d; }
  v *= (1.f / HH);
  float inv = rsqrtf(v + 1e-5f);
  for (int c = 0; c < HH; ++c) {
    float o = (row[c] - m) * inv * g[c] + b[c];
    dst[c] = o;
    dsth[c] = (_Float16)o;
  }
}

__global__ __launch_bounds__(256) void layer_kernel(LayerArgs A) {
  // union region: edge phase { E0 E1 M1 } vs node phase { bufA bufB Ast Ast2 }
  // buffers are always derived from `uni` by CONSTANT/selected byte offsets so the
  // compiler can infer addrspace(3) and emit ds_load_b128 (not flat) for fragments.
  __shared__ __align__(16) char uni[45056];
  __shared__ float hf32[16 * HH];
  __shared__ float aggs[16 * HH];
  __shared__ float pmvs[16 * MVC];
  __shared__ float mmvs[MVC];
  __shared__ int   nbrs_all[256];
  __shared__ float envv_all[256];

  _Float16* M1   = (_Float16*)(uni + 36864); // [32][128]
  float*    bufA = (float*)uni;              // [16][128]
  float*    bufB = (float*)(uni + 8192);     // [16][128]
  _Float16* Ast  = (_Float16*)(uni + 16384); // [16][256]
  _Float16* Ast2 = (_Float16*)(uni + 24576); // [16][256]

  int tid  = threadIdx.x;
  int lane = tid & 31;
  int w    = tid >> 5;
  int base = blockIdx.x * 16;
  int bidx = base >> 11;
  int col  = lane & 15;
  int rb   = (lane >> 4) * 8;

  // prefetch this layer's weight panels (global_prefetch_b8)
  __builtin_prefetch(A.w1t + (size_t)tid * 144, 0, 0);
  __builtin_prefetch(A.w2t + (size_t)tid * 64, 0, 0);
  __builtin_prefetch(A.fut + (size_t)tid * 128, 0, 0);

  // h tile (f32), agg init, neighbor table for all 256 edges of this block
  for (int q = tid; q < 16 * HH; q += 256) {
    hf32[q] = A.h_in[(size_t)(base + (q >> 7)) * HH + (q & 127)];
    aggs[q] = 0.f;
  }
  {
    int ln = tid >> 4, e = tid & 15;
    int grow = base + ln;
    nbrs_all[tid] = (bidx << 11) + A.idx[grow * KK + e];
    envv_all[tid] = A.env[grow * KK + e];
  }
  if (tid < MVC) mmvs[tid] = A.mmv[bidx * MVC + tid];
  __syncthreads();

  // async staging of one edge chunk (2 nodes = 32 rows x 288 halfs) into uni+ebyte
  // 36 x 16-byte segments per row: 16 h_i, 16 h_j, 4 rbf
  auto issue_chunk = [&](int c, int ebyte) {
    _Float16* E = (_Float16*)(uni + ebyte);
    for (int q = tid; q < 32 * 36; q += 256) {
      int r = q & 31, s = q >> 5;
      int ln = c * 2 + (r >> 4), e = r & 15;
      const _Float16* src;
      int dcol;
      if (s < 16)      { src = A.h16_in + (size_t)(base + ln) * HH + s * 8;                     dcol = s * 8; }
      else if (s < 32) { src = A.h16_in + (size_t)nbrs_all[c * 32 + r] * HH + (s - 16) * 8;     dcol = 128 + (s - 16) * 8; }
      else             { src = A.rbf16 + ((size_t)(base + ln) * KK + e) * 32 + (s - 32) * 8;    dcol = 256 + (s - 32) * 8; }
      unsigned lds_off = (unsigned)(size_t)(void*)(E + r * 288 + dcol);
      async_cp16(lds_off, (const void*)src);
    }
  };

  issue_chunk(0, 0);

  // ===== edge phase: 8 pipelined chunks =====
  for (int c = 0; c < 8; ++c) {
    int ebyte = (c & 1) * 18432;
    async_wait0();        // chunk c landed in LDS
    __syncthreads();      // visible to all waves; prior GEMMs done
    if (c < 7) issue_chunk(c + 1, ebyte ^ 18432);   // overlap next gather with compute
    const _Float16* E = (const _Float16*)(uni + ebyte);

    // GEMM1: [32 x 288] @ [288 x 128] -> silu -> M1 (f16)
    for (int t = w; t < 16; t += 8) {
      int mt = t >> 3, nt = t & 7;
      v8f acc = {};
      for (int kb = 0; kb < 288; kb += 32) {
        v16h a  = load_a16(E + mt * 16 * 288, 288, kb);
        v16h bb = load_bT(A.w1t, 288, nt * 16, kb);
        acc = __builtin_amdgcn_wmma_f32_16x16x32_f16(false, a, false, bb, (short)0, acc, false, false);
      }
#pragma unroll
      for (int r = 0; r < 8; ++r) {
        int rr = mt * 16 + rb + r, n = nt * 16 + col;
        M1[rr * HH + n] = (_Float16)siluf(acc[r] + A.e1b[n]);
      }
    }
    __syncthreads();
    // GEMM2: [32 x 128] @ [128 x 128] -> env scale -> mean over 16 edges -> agg
    for (int t = w; t < 16; t += 8) {
      int mt = t >> 3, nt = t & 7;
      v8f acc = {};
      for (int kb = 0; kb < 128; kb += 32) {
        v16h a  = load_a16(M1 + mt * 16 * HH, HH, kb);
        v16h bb = load_bT(A.w2t, 128, nt * 16, kb);
        acc = __builtin_amdgcn_wmma_f32_16x16x32_f16(false, a, false, bb, (short)0, acc, false, false);
      }
      float s = 0.f;
#pragma unroll
      for (int r = 0; r < 8; ++r) {
        int er = mt * 16 + rb + r;
        s += (acc[r] + A.e2b[nt * 16 + col]) * envv_all[c * 32 + er];
      }
      s += __shfl_xor(s, 16);
      if (lane < 16) aggs[(c * 2 + mt) * HH + nt * 16 + lane] += s * (1.f / 16.f);
    }
  }
  __syncthreads();

  // ===== node phase (M = 16 nodes) =====
  for (int q = tid; q < 16 * 256; q += 256) {
    int r = q >> 8, cc = q & 255;
    float v = (cc < 128) ? hf32[r * HH + cc] : aggs[r * HH + (cc - 128)];
    Ast[q] = (_Float16)v;
  }
  __syncthreads();
  {
    int nt = w;
    v8f acc = {};
    for (int kb = 0; kb < 256; kb += 32) {
      v16h a  = load_a16(Ast, 256, kb);
      v16h bb = load_bT(A.nmt, 256, nt * 16, kb);
      acc = __builtin_amdgcn_wmma_f32_16x16x32_f16(false, a, false, bb, (short)0, acc, false, false);
    }
#pragma unroll
    for (int r = 0; r < 8; ++r) {
      int rr = rb + r, n = nt * 16 + col;
      bufA[rr * HH + n] = hf32[rr * HH + n] + acc[r] + A.nmb[n];
    }
  }
  __syncthreads();
  if (tid < 16) ln_row_f32(bufA + tid * HH, A.lng, A.lnb);  // local
  if (tid < 128) {
    int r = tid >> 3, j = tid & 7;
    float a = A.ppb[j];
    for (int k = 0; k < HH; ++k) a += hf32[r * HH + k] * A.ppw[k * MVC + j];
    pmvs[r * MVC + j] = a;
  }
  __syncthreads();
  for (int q = tid; q < 16 * 64; q += 256) {
    int r = q >> 6, uv = q & 63;
    Ast[q] = (_Float16)(pmvs[r * MVC + (uv >> 3)] * mmvs[uv & 7]);
  }
  __syncthreads();
  {
    int nt = w;
    v8f acc = {};
    for (int kb = 0; kb < 64; kb += 32) {
      v16h a  = load_a16(Ast, 64, kb);
      v16h bb = load_bT(A.ipt, 64, nt * 16, kb);
      acc = __builtin_amdgcn_wmma_f32_16x16x32_f16(false, a, false, bb, (short)0, acc, false, false);
    }
#pragma unroll
    for (int r = 0; r < 8; ++r) {
      int rr = rb + r, n = nt * 16 + col;
      bufB[rr * HH + n] = hf32[rr * HH + n] + 0.1f * (acc[r] + A.ipb[n]);
    }
  }
  __syncthreads();
  if (tid < 16) ln_row_f32(bufB + tid * HH, A.gng, A.gnb);  // glob
  __syncthreads();
  for (int q = tid; q < 16 * 256; q += 256) {
    int r = q >> 8, cc = q & 255;
    float v = (cc < 128) ? bufA[r * HH + cc] : bufB[r * HH + (cc - 128)];
    Ast[q] = (_Float16)v;
  }
  __syncthreads();
  {
    int nt = w;
    v8f acc = {};
    for (int kb = 0; kb < 256; kb += 32) {
      v16h a  = load_a16(Ast, 256, kb);
      v16h bb = load_bT(A.fut, 256, nt * 16, kb);
      acc = __builtin_amdgcn_wmma_f32_16x16x32_f16(false, a, false, bb, (short)0, acc, false, false);
    }
#pragma unroll
    for (int r = 0; r < 8; ++r) {
      int rr = rb + r, n = nt * 16 + col;
      bufA[rr * HH + n] = geluf(acc[r] + A.fub[n]);   // fused
    }
  }
  __syncthreads();
  if (tid < 16) ln_row_to_h(bufA + tid * HH, Ast + tid * HH, A.n1g, A.n1b);
  __syncthreads();
  for (int t = w; t < 16; t += 8) {
    int nt = t;
    v8f acc = {};
    for (int kb = 0; kb < 128; kb += 32) {
      v16h a  = load_a16(Ast, 128, kb);
      v16h bb = load_bT(A.fc1t, 128, nt * 16, kb);
      acc = __builtin_amdgcn_wmma_f32_16x16x32_f16(false, a, false, bb, (short)0, acc, false, false);
    }
#pragma unroll
    for (int r = 0; r < 8; ++r) {
      int rr = rb + r, n = nt * 16 + col;
      Ast2[rr * 256 + n] = (_Float16)geluf(acc[r] + A.fc1b[n]);
    }
  }
  __syncthreads();
  {
    int nt = w;
    v8f acc = {};
    for (int kb = 0; kb < 256; kb += 32) {
      v16h a  = load_a16(Ast2, 256, kb);
      v16h bb = load_bT(A.fc2t, 256, nt * 16, kb);
      acc = __builtin_amdgcn_wmma_f32_16x16x32_f16(false, a, false, bb, (short)0, acc, false, false);
    }
#pragma unroll
    for (int r = 0; r < 8; ++r) {
      int rr = rb + r, n = nt * 16 + col;
      bufB[rr * HH + n] = acc[r] + A.fc2b[n] + bufA[rr * HH + n];
    }
  }
  __syncthreads();
  for (int q = tid; q < 16 * HH; q += 256) {
    int r = q >> 7, cc = q & 127, g = cc >> 3, j = cc & 7;
    const float* yrow = bufB + r * HH + g * GSZ;
    float a = A.gmb[j];
#pragma unroll
    for (int u = 0; u < GSZ; ++u)
#pragma unroll
      for (int v2 = 0; v2 < GSZ; ++v2)
        a += yrow[u] * yrow[v2] * A.gmw[(u * GSZ + v2) * GSZ + j];
    bufA[q] = a;
  }
  __syncthreads();
  for (int q = tid; q < 16 * HH; q += 256) bufB[q] += 0.1f * bufA[q];
  __syncthreads();
  if (tid < 16)
    ln_row_out(bufB + tid * HH, A.h_out + (size_t)(base + tid) * HH,
               A.h16_out + (size_t)(base + tid) * HH, A.n2g, A.n2b);
}

// ---------------- output head ----------------
__global__ void out_kernel(const float* __restrict__ x, const float* __restrict__ h,
                           const float* __restrict__ out_w, const float* __restrict__ out_b,
                           float* __restrict__ out) {
  int q = blockIdx.x * 256 + threadIdx.x;
  if (q >= ROWS * 6) return;
  int row = q / 6, c = q % 6;
  float a = out_b[c];
  for (int k = 0; k < HH; ++k) a += h[(size_t)row * HH + k] * out_w[k * 6 + c];
  out[q] = x[q] + a;
}

extern "C" void kernel_launch(void* const* d_in, const int* in_sizes, int n_in,
                              void* d_out, int out_size, void* d_ws, size_t ws_size,
                              hipStream_t stream) {
  (void)in_sizes; (void)n_in; (void)out_size; (void)ws_size;
  const float* x     = (const float*)d_in[0];
  const float* in_w  = (const float*)d_in[1];
  const float* in_b  = (const float*)d_in[2];
  const float* e1_w  = (const float*)d_in[3];
  const float* e1_b  = (const float*)d_in[4];
  const float* e2_w  = (const float*)d_in[5];
  const float* e2_b  = (const float*)d_in[6];
  const float* nm_w  = (const float*)d_in[7];
  const float* nm_b  = (const float*)d_in[8];
  const float* ln_g  = (const float*)d_in[9];
  const float* ln_b  = (const float*)d_in[10];
  const float* pp_w  = (const float*)d_in[11];
  const float* pp_b  = (const float*)d_in[12];
  const float* mp_w  = (const float*)d_in[13];
  const float* mp_b  = (const float*)d_in[14];
  const float* ip_w  = (const float*)d_in[15];
  const float* ip_b  = (const float*)d_in[16];
  const float* gn_g  = (const float*)d_in[17];
  const float* gn_b  = (const float*)d_in[18];
  const float* fu_w  = (const float*)d_in[19];
  const float* fu_b  = (const float*)d_in[20];
  const float* fc1_w = (const float*)d_in[21];
  const float* fc1_b = (const float*)d_in[22];
  const float* fc2_w = (const float*)d_in[23];
  const float* fc2_b = (const float*)d_in[24];
  const float* gm_w  = (const float*)d_in[25];
  const float* gm_b  = (const float*)d_in[26];
  const float* n1_g  = (const float*)d_in[27];
  const float* n1_b  = (const float*)d_in[28];
  const float* n2_g  = (const float*)d_in[29];
  const float* n2_b  = (const float*)d_in[30];
  const float* ow    = (const float*)d_in[31];
  const float* ob    = (const float*)d_in[32];

  char* wp = (char*)d_ws;
  auto alloc = [&](size_t bytes) -> void* {
    void* p = (void*)wp;
    wp += (bytes + 255) & ~(size_t)255;
    return p;
  };
  int*      idxb  = (int*)alloc((size_t)ROWS * KK * sizeof(int));
  _Float16* rbf16 = (_Float16*)alloc((size_t)ROWS * KK * 32 * 2);
  float*    envb  = (float*)alloc((size_t)ROWS * KK * sizeof(float));
  float*    h0    = (float*)alloc((size_t)ROWS * HH * sizeof(float));
  float*    h1    = (float*)alloc((size_t)ROWS * HH * sizeof(float));
  _Float16* h16_0 = (_Float16*)alloc((size_t)ROWS * HH * 2);
  _Float16* h16_1 = (_Float16*)alloc((size_t)ROWS * HH * 2);
  float*    mmvb  = (float*)alloc((size_t)BB * MVC * sizeof(float));
  _Float16 *w1t[LL], *w2t[LL], *nmt[LL], *ipt[LL], *fut[LL], *fc1t[LL], *fc2t[LL];
  for (int i = 0; i < LL; ++i) {
    w1t[i]  = (_Float16*)alloc((size_t)128 * 288 * 2);
    w2t[i]  = (_Float16*)alloc((size_t)128 * 128 * 2);
    nmt[i]  = (_Float16*)alloc((size_t)128 * 256 * 2);
    ipt[i]  = (_Float16*)alloc((size_t)128 * 64  * 2);
    fut[i]  = (_Float16*)alloc((size_t)128 * 256 * 2);
    fc1t[i] = (_Float16*)alloc((size_t)256 * 128 * 2);
    fc2t[i] = (_Float16*)alloc((size_t)128 * 256 * 2);
  }

  auto prep = [&](const float* s, _Float16* d, int K, int N, int Kpad) {
    int tot = N * Kpad;
    prep_kernel<<<(tot + 255) / 256, 256, 0, stream>>>(s, d, K, N, Kpad);
  };
  for (int i = 0; i < LL; ++i) {
    prep(e1_w  + (size_t)i * 276 * 128, w1t[i],  276, 128, 288);
    prep(e2_w  + (size_t)i * 128 * 128, w2t[i],  128, 128, 128);
    prep(nm_w  + (size_t)i * 256 * 128, nmt[i],  256, 128, 256);
    prep(ip_w  + (size_t)i * 64  * 128, ipt[i],  64,  128, 64);
    prep(fu_w  + (size_t)i * 256 * 128, fut[i],  256, 128, 256);
    prep(fc1_w + (size_t)i * 128 * 256, fc1t[i], 128, 256, 128);
    prep(fc2_w + (size_t)i * 256 * 128, fc2t[i], 256, 128, 256);
  }

  knn_kernel<<<ROWS, 32, 0, stream>>>(x, idxb, rbf16, envb);
  embed_kernel<<<(ROWS * HH) / 256, 256, 0, stream>>>(x, in_w, in_b, h0, h16_0);

  float* hin = h0;       float* hout = h1;
  _Float16* h16in = h16_0; _Float16* h16out = h16_1;
  for (int i = 0; i < LL; ++i) {
    meanfield_kernel<<<BB, 128, 0, stream>>>(hin, mp_w + (size_t)i * HH * MVC,
                                             mp_b + (size_t)i * MVC, mmvb);
    LayerArgs la;
    la.h_in = hin; la.h_out = hout;
    la.h16_in = h16in; la.h16_out = h16out;
    la.idx = idxb; la.rbf16 = rbf16; la.env = envb; la.mmv = mmvb;
    la.w1t = w1t[i]; la.w2t = w2t[i]; la.nmt = nmt[i]; la.ipt = ipt[i];
    la.fut = fut[i]; la.fc1t = fc1t[i]; la.fc2t = fc2t[i];
    la.e1b = e1_b + (size_t)i * HH;  la.e2b = e2_b + (size_t)i * HH;
    la.nmb = nm_b + (size_t)i * HH;
    la.lng = ln_g + (size_t)i * HH;  la.lnb = ln_b + (size_t)i * HH;
    la.ppw = pp_w + (size_t)i * HH * MVC; la.ppb = pp_b + (size_t)i * MVC;
    la.ipb = ip_b + (size_t)i * HH;
    la.gng = gn_g + (size_t)i * HH;  la.gnb = gn_b + (size_t)i * HH;
    la.fub = fu_b + (size_t)i * HH;
    la.fc1b = fc1_b + (size_t)i * 2 * HH; la.fc2b = fc2_b + (size_t)i * HH;
    la.gmw = gm_w + (size_t)i * 64 * 8; la.gmb = gm_b + (size_t)i * 8;
    la.n1g = n1_g + (size_t)i * HH;  la.n1b = n1_b + (size_t)i * HH;
    la.n2g = n2_g + (size_t)i * HH;  la.n2b = n2_b + (size_t)i * HH;
    layer_kernel<<<ROWS / 16, 256, 0, stream>>>(la);
    float* t = hin; hin = hout; hout = t;
    _Float16* t2 = h16in; h16in = h16out; h16out = t2;
  }
  out_kernel<<<(ROWS * 6 + 255) / 256, 256, 0, stream>>>(x, hin, ow, ob, (float*)d_out);
}
